// RSSM_91018946937098
// MI455X (gfx1250) — matrix operand order repaired and checked by reading, not compiled, not run on hardware
//
#include <hip/hip_runtime.h>

// ---------------- problem constants ----------------
#define B_      64
#define T_      64
#define EMBED_  1536
#define ACT_    6
#define D_      1024      // DETER == HID == STOCH*CLASSES
#define SC_     1024
#define GINP_   1056      // GRU_IN (1030) padded up to a multiple of 32
#define POSTK_  2560      // DETER + EMBED

typedef __bf16 bf16;
typedef __attribute__((ext_vector_type(16))) __bf16 v16bf;
typedef __attribute__((ext_vector_type(8)))  __bf16 v8bf;
typedef __attribute__((ext_vector_type(8)))  float  v8f;
typedef __attribute__((ext_vector_type(4)))  unsigned int v4u;
typedef __attribute__((ext_vector_type(8)))  int v8i;
typedef __attribute__((ext_vector_type(4)))  int v4i;

static __device__ __forceinline__ v16bf frag16(v8bf lo, v8bf hi) {
  return __builtin_shufflevector(lo, hi, 0,1,2,3,4,5,6,7,8,9,10,11,12,13,14,15);
}

// A fragment (16x32 bf16, 16-bit A-matrix layout, ISA 7.12.2):
// lane L: M = L&15 ; kb = (L>=16)?8:0 ; elems 0..7 -> K=k0+kb.. ; elems 8..15 -> K=k0+16+kb..
static __device__ __forceinline__ v16bf load_a(const bf16* row, int k0, int lane) {
  const int kb = (lane >> 4) << 3;
  v8bf lo = *reinterpret_cast<const v8bf*>(row + k0 + kb);
  v8bf hi = *reinterpret_cast<const v8bf*>(row + k0 + kb + 16);
  return frag16(lo, hi);
}
// B fragment (32x16 bf16; W stored row-major as W[o][k], o = column of the tile):
// lane L: kb = (L>=16)?8:0 ; elems 0..7 -> K=k0+kb.. ; elems 8..15 -> K=k0+16+kb.. (same as A)
static __device__ __forceinline__ v16bf load_b(const bf16* __restrict__ row, int k0, int lane) {
  const int kb = (lane >> 4) << 3;
  v8bf lo = *reinterpret_cast<const v8bf*>(row + k0 + kb);
  v8bf hi = *reinterpret_cast<const v8bf*>(row + k0 + kb + 16);
  return frag16(lo, hi);
}
static __device__ __forceinline__ v8f wmma_bf(v16bf a, v16bf b, v8f c) {
  return __builtin_amdgcn_wmma_f32_16x16x32_bf16(false, a, false, b, (short)0, c, false, false);
}
static __device__ __forceinline__ float sigmoidf_(float x) { return 1.f / (1.f + expf(-x)); }

// ---------------- Tensor Data Mover: 2D tile (rows x tile_k bf16) -> LDS ----------------
// D# per CDNA5 ISA ch.8: group0 = {count/type, lds_addr, global_addr}, group1 = dims/strides.
static __device__ __forceinline__ void tdm_load_2d(bf16* lds_dst, const bf16* gsrc,
                                                   int dim0, int rows, int tile_k) {
  const unsigned long long ga = (unsigned long long)(const void*)gsrc;
  const unsigned int laddr = (unsigned int)(unsigned long long)(void*)lds_dst;
  v4u g0;
  g0[0] = 1u;                                                     // count=1 (valid user D#)
  g0[1] = laddr;                                                  // lds_addr (bytes)
  g0[2] = (unsigned int)ga;                                       // global_addr[31:0]
  g0[3] = (unsigned int)((ga >> 32) & 0x01FFFFFFull) | (2u << 30);// global_addr[56:32] | type=2
  v8i g1;
  g1[0] = 0x10000;                                 // workgroup_mask=0, data_size=1 (2 bytes)
  g1[1] = (int)(((unsigned)dim0 & 0xFFFFu) << 16); // tensor_dim0[15:0] @ bits 63:48
  g1[2] = (int)(((unsigned)rows << 16) | ((unsigned)dim0 >> 16)); // dim0[31:16] | tensor_dim1[15:0]
  g1[3] = (int)((unsigned)tile_k << 16);           // tile_dim0 @ bits 127:112
  g1[4] = rows;                                    // tile_dim1
  g1[5] = dim0;                                    // tensor_dim0_stride[31:0] (row pitch)
  g1[6] = 0;
  g1[7] = 0;
  const v4i gz = {0, 0, 0, 0};
#if __clang_major__ >= 23
  const v8i gz8 = {0, 0, 0, 0, 0, 0, 0, 0};
  __builtin_amdgcn_tensor_load_to_lds(g0, g1, gz, gz, gz8, 0);
#else
  __builtin_amdgcn_tensor_load_to_lds(g0, g1, gz, gz, 0);
#endif
}

// ---------------- weight fp32 -> bf16 (with K padding) ----------------
__global__ void k_convert(const float* __restrict__ src, bf16* __restrict__ dst,
                          int O, int K, int Kpad) {
  const long n = (long)O * Kpad;
  for (long i = (long)blockIdx.x * blockDim.x + threadIdx.x; i < n;
       i += (long)gridDim.x * blockDim.x) {
    const int o = (int)(i / Kpad), k = (int)(i - (long)o * Kpad);
    dst[i] = (k < K) ? (bf16)src[(long)o * K + k] : (bf16)0.f;
  }
}

__global__ void k_zero(float* __restrict__ a, float* __restrict__ b, int n) {
  for (int i = blockIdx.x * blockDim.x + threadIdx.x; i < n; i += gridDim.x * blockDim.x) {
    a[i] = 0.f; b[i] = 0.f;
  }
}

// ---------------- per-timestep: reset + bf16 staging ----------------
__global__ void k_prep(const float* __restrict__ h, const float* __restrict__ z,
                       const float* __restrict__ actions, const float* __restrict__ embeds,
                       const int* __restrict__ is_first, int t,
                       float* __restrict__ hkeep, bf16* __restrict__ hbf,
                       bf16* __restrict__ gin, bf16* __restrict__ embt) {
  const int N1 = B_ * GINP_, N2 = B_ * D_, N3 = B_ * EMBED_;
  for (int i = blockIdx.x * blockDim.x + threadIdx.x; i < N1 + N2 + N3;
       i += gridDim.x * blockDim.x) {
    if (i < N1) {
      const int m = i / GINP_, k = i - m * GINP_;
      const float keep = 1.f - (float)is_first[m * T_ + t];
      float v = 0.f;
      if (k < SC_)             v = z[m * SC_ + k] * keep;
      else if (k < SC_ + ACT_) v = actions[(m * T_ + t) * ACT_ + (k - SC_)];
      gin[i] = (bf16)v;
    } else if (i < N1 + N2) {
      const int j = i - N1, m = j / D_;
      const float keep = 1.f - (float)is_first[m * T_ + t];
      const float hv = h[j] * keep;
      hkeep[j] = hv;
      hbf[j] = (bf16)hv;
    } else {
      const int j = i - N1 - N2, m = j / EMBED_, k = j - m * EMBED_;
      embt[j] = (bf16)embeds[(m * T_ + t) * EMBED_ + k];
    }
  }
}

// ---------------- GRU step: TDM A-slab -> LDS, fused gx+gh WMMA + gate math ----------------
// grid = 16 workgroups, one per 64-column slice of h. 8 waves:
// wave pair s=0 -> r gate (Wih+Whh fused), s=1 -> z gate (fused),
// s=2 -> nx (Wih only), s=3 -> nh (Whh only).
// The 64x32 bf16 A-slab (4 KB) is DMA'd into LDS by the TDM once per K-step and
// aliases the first rows of g4 (g4 is only live after the K loops).
__global__ __launch_bounds__(256) void k_gru(
    const bf16* __restrict__ gin, const bf16* __restrict__ hbf,
    const float* __restrict__ hkeep,
    const bf16* __restrict__ Wih, const bf16* __restrict__ Whh,
    const float* __restrict__ bih, const float* __restrict__ bhh,
    float* __restrict__ hout, bf16* __restrict__ hnewbf,
    float* __restrict__ out_latent, float* __restrict__ out_hseq, int t) {
  __shared__ float g4[4][64][64];   // 64 KB: r, z, nx, nh slices (tail of K loops reuses head)
  bf16* slab = reinterpret_cast<bf16*>(&g4[0][0][0]);   // 64 rows x 32 cols bf16 = 4 KB
  const int tid = threadIdx.x, w = tid >> 5, lane = tid & 31, ln = lane & 15;
  const int s = w >> 1, ntb = (w & 1) * 2, c0 = blockIdx.x * 64;
  const v8f Z8 = {0.f,0.f,0.f,0.f,0.f,0.f,0.f,0.f};
  v8f acc[4][2];
  #pragma unroll
  for (int mt = 0; mt < 4; mt++)
    #pragma unroll
    for (int j = 0; j < 2; j++) acc[mt][j] = Z8;

  // phase 1: A = gru_in (K = GINP_), waves s=0,1,2 accumulate vs W_ih
  for (int k0 = 0; k0 < GINP_; k0 += 32) {
    __syncthreads();                       // previous slab fully consumed
    if (tid == 0) {
      tdm_load_2d(slab, gin + k0, GINP_, 64, 32);
      __builtin_amdgcn_s_wait_tensorcnt(0);
    }
    __syncthreads();                       // slab ready for all waves
    if (s != 3) {
      const int go = s * D_;
      v16bf a[4];
      #pragma unroll
      for (int mt = 0; mt < 4; mt++) a[mt] = load_a(slab + (mt * 16 + ln) * 32, 0, lane);
      #pragma unroll
      for (int j = 0; j < 2; j++) {
        const bf16* brow = Wih + (go + c0 + (ntb + j) * 16 + ln) * GINP_;
        __builtin_prefetch(brow + k0 + 128, 0, 3);
        v16bf b = load_b(brow, k0, lane);
        #pragma unroll
        for (int mt = 0; mt < 4; mt++) acc[mt][j] = wmma_bf(a[mt], b, acc[mt][j]);
      }
    }
  }
  // phase 2: A = h (K = D_), waves s=0,1,3 accumulate vs W_hh
  for (int k0 = 0; k0 < D_; k0 += 32) {
    __syncthreads();
    if (tid == 0) {
      tdm_load_2d(slab, hbf + k0, D_, 64, 32);
      __builtin_amdgcn_s_wait_tensorcnt(0);
    }
    __syncthreads();
    if (s != 2) {
      const int go = (s == 3) ? 2 * D_ : s * D_;
      v16bf a[4];
      #pragma unroll
      for (int mt = 0; mt < 4; mt++) a[mt] = load_a(slab + (mt * 16 + ln) * 32, 0, lane);
      #pragma unroll
      for (int j = 0; j < 2; j++) {
        const bf16* brow = Whh + (go + c0 + (ntb + j) * 16 + ln) * D_;
        __builtin_prefetch(brow + k0 + 128, 0, 3);
        v16bf b = load_b(brow, k0, lane);
        #pragma unroll
        for (int mt = 0; mt < 4; mt++) acc[mt][j] = wmma_bf(a[mt], b, acc[mt][j]);
      }
    }
  }
  __syncthreads();   // slab dead; g4 storage may now be written

  // bias + stage to LDS (C/D layout: lane L -> N=L&15, elem v -> M = v + 8*(L>=16))
  #pragma unroll
  for (int j = 0; j < 2; j++) {
    const int c = c0 + (ntb + j) * 16 + ln;
    float bias;
    if (s == 0)      bias = bih[c] + bhh[c];
    else if (s == 1) bias = bih[D_ + c] + bhh[D_ + c];
    else if (s == 2) bias = bih[2 * D_ + c];
    else             bias = bhh[2 * D_ + c];
    #pragma unroll
    for (int mt = 0; mt < 4; mt++)
      #pragma unroll
      for (int v = 0; v < 8; v++)
        g4[s][mt * 16 + v + ((lane >> 4) << 3)][(ntb + j) * 16 + ln] = acc[mt][j][v] + bias;
  }
  __syncthreads();
  for (int i = tid; i < 64 * 64; i += 256) {
    const int m = i >> 6, n = i & 63, c = c0 + n;
    const float r  = sigmoidf_(g4[0][m][n]);
    const float u  = sigmoidf_(g4[1][m][n]);
    const float nn = tanhf(g4[2][m][n] + r * g4[3][m][n]);
    const float hk = hkeep[m * D_ + c];
    const float hn = (1.f - u) * nn + u * hk;
    hout[m * D_ + c] = hn;
    hnewbf[m * D_ + c] = (bf16)hn;
    out_latent[(m * T_ + t) * 2048 + c] = hn;
    out_hseq[(m * T_ + t) * D_ + c] = hn;
  }
}

// ---------------- MLP layer 1 + LayerNorm + SiLU ----------------
// grid = 8 workgroups: 0..3 prior (16 rows each), 4..7 post (16 rows each, K=2560).
__global__ __launch_bounds__(256) void k_mlp1(
    const bf16* __restrict__ hnew, const bf16* __restrict__ embt,
    const bf16* __restrict__ w1p, const float* __restrict__ b1p,
    const float* __restrict__ gp, const float* __restrict__ bep,
    const bf16* __restrict__ w1q, const float* __restrict__ b1q,
    const float* __restrict__ gq, const float* __restrict__ beq,
    bf16* __restrict__ hidp, bf16* __restrict__ hidq) {
  __shared__ float sh[16][1024];    // 64 KB staging for LayerNorm
  const int tid = threadIdx.x, w = tid >> 5, lane = tid & 31, ln = lane & 15;
  const bool post = blockIdx.x >= 4;
  const int m0 = (post ? (int)blockIdx.x - 4 : (int)blockIdx.x) * 16;
  const bf16* W = post ? w1q : w1p;
  const int ldb = post ? POSTK_ : D_;
  const int K = ldb;
  const bf16* arow_h = hnew + (m0 + ln) * D_;
  const bf16* arow_e = embt + (m0 + ln) * EMBED_ - D_;  // valid for k0 >= D_
  const v8f Z8 = {0.f,0.f,0.f,0.f,0.f,0.f,0.f,0.f};
  v8f acc[8];
  #pragma unroll
  for (int i = 0; i < 8; i++) acc[i] = Z8;

  for (int k0 = 0; k0 < K; k0 += 32) {
    const bf16* arow = (k0 < D_) ? arow_h : arow_e;
    v16bf a = load_a(arow, k0, lane);
    #pragma unroll
    for (int i = 0; i < 8; i++) {
      const bf16* brow = W + ((w * 8 + i) * 16 + ln) * ldb;
      v16bf b = load_b(brow, k0, lane);
      acc[i] = wmma_bf(a, b, acc[i]);
    }
  }
  const float* b1 = post ? b1q : b1p;
  #pragma unroll
  for (int i = 0; i < 8; i++) {
    const int col = (w * 8 + i) * 16 + ln;
    const float bias = b1[col];
    #pragma unroll
    for (int v = 0; v < 8; v++)
      sh[v + ((lane >> 4) << 3)][col] = acc[i][v] + bias;
  }
  __syncthreads();
  const float* gg = post ? gq : gp;
  const float* be = post ? beq : bep;
  bf16* out = post ? hidq : hidp;
  #pragma unroll
  for (int rr = 0; rr < 2; rr++) {
    const int r = w * 2 + rr;
    float sum = 0.f, sq = 0.f;
    for (int j = 0; j < 32; j++) { const float x = sh[r][lane + 32 * j]; sum += x; sq += x * x; }
    #pragma unroll
    for (int o = 16; o >= 1; o >>= 1) { sum += __shfl_xor(sum, o); sq += __shfl_xor(sq, o); }
    const float mean = sum * (1.f / 1024.f);
    const float var  = sq * (1.f / 1024.f) - mean * mean;
    const float rs   = rsqrtf(var + 1e-5f);
    for (int j = 0; j < 32; j++) {
      const int c = lane + 32 * j;
      const float x = sh[r][c];
      const float y = (x - mean) * rs * gg[c] + be[c];
      out[(m0 + r) * D_ + c] = (bf16)(y * sigmoidf_(y));
    }
  }
}

// ---------------- MLP layer 2 + straight-through sample ----------------
// grid = 8 workgroups: 0..3 prior logits, 4..7 post logits + z sample.
__global__ __launch_bounds__(256) void k_mlp2(
    const bf16* __restrict__ hidp, const bf16* __restrict__ hidq,
    const bf16* __restrict__ w2p, const float* __restrict__ b2p,
    const bf16* __restrict__ w2q, const float* __restrict__ b2q,
    float* __restrict__ out_prior, float* __restrict__ out_post,
    float* __restrict__ out_latent, float* __restrict__ zws, int t) {
  __shared__ float sh[16][1024];
  const int tid = threadIdx.x, w = tid >> 5, lane = tid & 31, ln = lane & 15;
  const bool post = blockIdx.x >= 4;
  const int m0 = (post ? (int)blockIdx.x - 4 : (int)blockIdx.x) * 16;
  const bf16* A = (post ? hidq : hidp) + (m0 + ln) * D_;
  const bf16* W = post ? w2q : w2p;
  const float* b2 = post ? b2q : b2p;
  const v8f Z8 = {0.f,0.f,0.f,0.f,0.f,0.f,0.f,0.f};
  v8f acc[8];
  #pragma unroll
  for (int i = 0; i < 8; i++) acc[i] = Z8;

  for (int k0 = 0; k0 < D_; k0 += 32) {
    v16bf a = load_a(A, k0, lane);
    #pragma unroll
    for (int i = 0; i < 8; i++) {
      v16bf b = load_b(W + ((w * 8 + i) * 16 + ln) * D_, k0, lane);
      acc[i] = wmma_bf(a, b, acc[i]);
    }
  }
  float* outlog = post ? out_post : out_prior;
  #pragma unroll
  for (int i = 0; i < 8; i++) {
    const int col = (w * 8 + i) * 16 + ln;
    const float bias = b2[col];
    #pragma unroll
    for (int v = 0; v < 8; v++) {
      const int mloc = v + ((lane >> 4) << 3);
      const float val = acc[i][v] + bias;
      outlog[((m0 + mloc) * T_ + t) * SC_ + col] = val;
      if (post) sh[mloc][col] = val;
    }
  }
  if (post) {
    __syncthreads();
    // 16 rows x 32 groups of 32 classes -> 512 softmax/argmax tasks
    for (int p = tid; p < 512; p += 256) {
      const int r = p >> 5, g = p & 31, bb = m0 + r;
      const float* row = &sh[r][g * 32];
      float mx = row[0]; int am = 0;
      #pragma unroll
      for (int c = 1; c < 32; c++) { const float l = row[c]; if (l > mx) { mx = l; am = c; } }
      float S = 0.f;
      for (int c = 0; c < 32; c++) S += expf(row[c] - mx);
      const float inv = 1.f / S;
      for (int c = 0; c < 32; c++) {
        const float pcl = expf(row[c] - mx) * inv;
        const float oh = (c == am) ? 1.f : 0.f;
        const float zc = (oh - pcl) + pcl;   // straight-through value, same fp ops as ref
        zws[bb * SC_ + g * 32 + c] = zc;
        out_latent[(bb * T_ + t) * 2048 + D_ + g * 32 + c] = zc;
      }
    }
  }
}

// ---------------- host launcher ----------------
extern "C" void kernel_launch(void* const* d_in, const int* in_sizes, int n_in,
                              void* d_out, int out_size, void* d_ws, size_t ws_size,
                              hipStream_t stream) {
  (void)in_sizes; (void)n_in; (void)out_size; (void)ws_size;
  const float* embeds   = (const float*)d_in[0];
  const float* actions  = (const float*)d_in[1];
  const int*   is_first = (const int*)d_in[2];
  const float* W_ih     = (const float*)d_in[3];
  const float* W_hh     = (const float*)d_in[4];
  const float* b_ih     = (const float*)d_in[5];
  const float* b_hh     = (const float*)d_in[6];
  const float* prior_w1 = (const float*)d_in[7];
  const float* prior_b1 = (const float*)d_in[8];
  const float* prior_g  = (const float*)d_in[9];
  const float* prior_be = (const float*)d_in[10];
  const float* prior_w2 = (const float*)d_in[11];
  const float* prior_b2 = (const float*)d_in[12];
  const float* post_w1  = (const float*)d_in[13];
  const float* post_b1  = (const float*)d_in[14];
  const float* post_g   = (const float*)d_in[15];
  const float* post_be  = (const float*)d_in[16];
  const float* post_w2  = (const float*)d_in[17];
  const float* post_b2  = (const float*)d_in[18];

  float* out_latent = (float*)d_out;                       // (B,T,2048)
  float* out_hseq   = out_latent + (size_t)B_ * T_ * 2048; // (B,T,1024)
  float* out_post   = out_hseq   + (size_t)B_ * T_ * D_;   // (B,T,32,32)
  float* out_prior  = out_post   + (size_t)B_ * T_ * SC_;  // (B,T,32,32)

  char* wsp = (char*)d_ws;
  size_t off = 0;
  auto carve = [&](size_t bytes) -> void* {
    void* p = wsp + off;
    off += (bytes + 255) & ~(size_t)255;
    return p;
  };
  bf16*  Wih_bf = (bf16*)carve((size_t)3 * D_ * GINP_ * 2);
  bf16*  Whh_bf = (bf16*)carve((size_t)3 * D_ * D_ * 2);
  bf16*  w1p    = (bf16*)carve((size_t)D_ * D_ * 2);
  bf16*  w2p    = (bf16*)carve((size_t)D_ * D_ * 2);
  bf16*  w1q    = (bf16*)carve((size_t)D_ * POSTK_ * 2);
  bf16*  w2q    = (bf16*)carve((size_t)D_ * D_ * 2);
  float* hbuf   = (float*)carve((size_t)B_ * D_ * 4);
  float* zbuf   = (float*)carve((size_t)B_ * SC_ * 4);
  float* hkeep  = (float*)carve((size_t)B_ * D_ * 4);
  bf16*  gin_bf = (bf16*)carve((size_t)B_ * GINP_ * 2);
  bf16*  h_bf   = (bf16*)carve((size_t)B_ * D_ * 2);
  bf16*  hnewbf = (bf16*)carve((size_t)B_ * D_ * 2);
  bf16*  embtbf = (bf16*)carve((size_t)B_ * EMBED_ * 2);
  bf16*  hidpbf = (bf16*)carve((size_t)B_ * D_ * 2);
  bf16*  hidqbf = (bf16*)carve((size_t)B_ * D_ * 2);

  auto blks = [](long n) { return (int)((n + 255) / 256); };
  k_convert<<<blks(3L * D_ * GINP_), 256, 0, stream>>>(W_ih, Wih_bf, 3 * D_, 1030, GINP_);
  k_convert<<<blks(3L * D_ * D_),    256, 0, stream>>>(W_hh, Whh_bf, 3 * D_, D_, D_);
  k_convert<<<blks((long)D_ * D_),   256, 0, stream>>>(prior_w1, w1p, D_, D_, D_);
  k_convert<<<blks((long)D_ * D_),   256, 0, stream>>>(prior_w2, w2p, D_, D_, D_);
  k_convert<<<blks((long)D_ * POSTK_), 256, 0, stream>>>(post_w1, w1q, D_, POSTK_, POSTK_);
  k_convert<<<blks((long)D_ * D_),   256, 0, stream>>>(post_w2, w2q, D_, D_, D_);
  k_zero<<<256, 256, 0, stream>>>(hbuf, zbuf, B_ * D_);

  const int prep_n = B_ * GINP_ + B_ * D_ + B_ * EMBED_;
  for (int t = 0; t < T_; t++) {
    k_prep<<<blks(prep_n), 256, 0, stream>>>(hbuf, zbuf, actions, embeds, is_first, t,
                                             hkeep, h_bf, gin_bf, embtbf);
    k_gru<<<16, 256, 0, stream>>>(gin_bf, h_bf, hkeep, Wih_bf, Whh_bf, b_ih, b_hh,
                                  hbuf, hnewbf, out_latent, out_hseq, t);
    k_mlp1<<<8, 256, 0, stream>>>(hnewbf, embtbf,
                                  w1p, prior_b1, prior_g, prior_be,
                                  w1q, post_b1, post_g, post_be,
                                  hidpbf, hidqbf);
    k_mlp2<<<8, 256, 0, stream>>>(hidpbf, hidqbf, w2p, prior_b2, w2q, post_b2,
                                  out_prior, out_post, out_latent, zbuf, t);
  }
}
